// MultiheadLatentAttention_15281493639832
// MI455X (gfx1250) — matrix operand order repaired and checked by reading, not compiled, not run on hardware
//
#include <hip/hip_runtime.h>
#include <math.h>

// MLA fallback-attention pipeline for MI455X (gfx1250, wave32).
// All GEMMs use v_wmma_f32_16x16x32_f16 (f16 inputs converted in-register
// from fp32 streams, f32 accumulation). Memory-bound design: ~536 MB of fp32
// traffic @ 23.3 TB/s (~23 us floor); f16 WMMA keeps compute far below that.

#define B_ 64
#define L_ 512
#define DQ_ 2048
#define H_ 128
#define KH_ 128
#define DC_ 512
#define DCQ_ 1536
#define R_ 64

typedef __attribute__((ext_vector_type(16))) _Float16 v16h;
typedef __attribute__((ext_vector_type(8)))  float    v8f;

// ---------------- WMMA fragment loaders (wave32; A/B use contiguous k = 16*(lane/16) + j) ------

// A-matrix 16x32 f16: lanes 0-15 -> M=lane, K 0..15;
// lanes 16-31 -> M=lane-16, K 16..31.
__device__ __forceinline__ v16h load_a_frag(const float* A, int lda, int kb, int lane) {
    const int m  = lane & 15;
    const int k0 = (lane & 16) ? 16 : 0;
    const float4* p = reinterpret_cast<const float4*>(A + (size_t)m * lda + kb + k0);
    float4 x0 = p[0], x1 = p[1];   // K = kb+k0 .. +7
    float4 x2 = p[2], x3 = p[3];   // K = kb+k0+8 .. +15
    v16h a;
    a[0]=(_Float16)x0.x; a[1]=(_Float16)x0.y; a[2]=(_Float16)x0.z; a[3]=(_Float16)x0.w;
    a[4]=(_Float16)x1.x; a[5]=(_Float16)x1.y; a[6]=(_Float16)x1.z; a[7]=(_Float16)x1.w;
    a[8]=(_Float16)x2.x; a[9]=(_Float16)x2.y; a[10]=(_Float16)x2.z; a[11]=(_Float16)x2.w;
    a[12]=(_Float16)x3.x; a[13]=(_Float16)x3.y; a[14]=(_Float16)x3.z; a[15]=(_Float16)x3.w;
    return a;
}

// B-matrix 32x16 f16, TN source (B stored [N, K], K contiguous):
// lane -> N=lane&15; halves j=0..15 -> K = (lane<16?0:16)+j. 16 contiguous floats.
__device__ __forceinline__ v16h load_b_tn(const float* Bm, int ldb, int kb, int lane) {
    const int n  = lane & 15;
    const int k0 = (lane & 16) ? 16 : 0;
    const float4* p = reinterpret_cast<const float4*>(Bm + (size_t)n * ldb + kb + k0);
    float4 x0 = p[0], x1 = p[1], x2 = p[2], x3 = p[3];
    v16h b;
    b[0]=(_Float16)x0.x; b[1]=(_Float16)x0.y; b[2]=(_Float16)x0.z; b[3]=(_Float16)x0.w;
    b[4]=(_Float16)x1.x; b[5]=(_Float16)x1.y; b[6]=(_Float16)x1.z; b[7]=(_Float16)x1.w;
    b[8]=(_Float16)x2.x; b[9]=(_Float16)x2.y; b[10]=(_Float16)x2.z; b[11]=(_Float16)x2.w;
    b[12]=(_Float16)x3.x; b[13]=(_Float16)x3.y; b[14]=(_Float16)x3.z; b[15]=(_Float16)x3.w;
    return b;
}

// B-matrix 32x16 f16, NN source (B stored [K, N], N contiguous): per-lane
// strided dword loads; for fixed j the 16 lanes hit a contiguous 64B segment.
__device__ __forceinline__ v16h load_b_nn(const float* Bm, int ldb, int kb, int lane) {
    const int n  = lane & 15;
    const int k0 = (lane & 16) ? 16 : 0;
    const float* p = Bm + (size_t)(kb + k0) * ldb + n;
    v16h b;
#pragma unroll
    for (int j = 0; j < 16; ++j) b[j] = (_Float16)p[(size_t)j * ldb];
    return b;
}

// C/D 16x16 f32: lane -> N=lane&15; VGPR v -> M = v + (lane<16?0:8).
__device__ __forceinline__ void store_c(float* C, int ldc, const v8f acc, int lane) {
    const int n    = lane & 15;
    const int moff = (lane & 16) ? 8 : 0;
#pragma unroll
    for (int v = 0; v < 8; ++v) C[(size_t)(moff + v) * ldc + n] = acc[v];
}

// --------------- generic wave-level GEMM: C[MT*16,16] = A x B ---------------
// A: [MT*16, K] row-major (K contiguous). B: TN -> [16, K]; NN -> [K, 16].
template<int MT, bool BTN>
__device__ __forceinline__ void wave_gemm(const float* A, int lda,
                                          const float* Bm, int ldb,
                                          float* C, int ldc, int K, int lane) {
    v8f acc[MT];
#pragma unroll
    for (int mt = 0; mt < MT; ++mt) { v8f z = {}; acc[mt] = z; }
    for (int kb = 0; kb < K; kb += 32) {
        v16h b = BTN ? load_b_tn(Bm, ldb, kb, lane) : load_b_nn(Bm, ldb, kb, lane);
#pragma unroll
        for (int mt = 0; mt < MT; ++mt) {
            v16h a = load_a_frag(A + (size_t)mt * 16 * lda, lda, kb, lane);
            acc[mt] = __builtin_amdgcn_wmma_f32_16x16x32_f16(
                false, a, false, b, (short)0, acc[mt], false, false);
        }
    }
#pragma unroll
    for (int mt = 0; mt < MT; ++mt) store_c(C + (size_t)mt * 16 * ldc, ldc, acc[mt], lane);
}

// ------------------------------- kernels ------------------------------------

// RoPE over kv_c -> kv (fp32), half-rotation form.
__global__ void k_rope(const float* __restrict__ kv_c, float* __restrict__ kv) {
    const int idx = blockIdx.x;                 // b*L + l
    const int l = idx & (L_ - 1);
    const float* src = kv_c + (size_t)idx * DC_;
    float* dst = kv + (size_t)idx * DC_;
    const float ln1e4_over = 9.210340371976184f / (float)DC_;  // ln(10000)/512
    for (int d = threadIdx.x; d < DC_; d += blockDim.x) {
        float x   = src[d];
        int   pr  = d >> 1;
        float inv = __expf(-(float)(2 * pr) * ln1e4_over);
        float th  = (float)l * inv;
        float s = sinf(th), c = cosf(th);
        float rh = (d < DC_ / 2) ? -src[d + DC_ / 2] : src[d - DC_ / 2];
        dst[d] = x * c + rh * s;
    }
}

// kv_sum[b,d] = sum_l kv[b,l,d]
__global__ void k_kvsum(const float* __restrict__ kv, float* __restrict__ kvsum) {
    const int b = blockIdx.x;
    const int d = blockIdx.y * blockDim.x + threadIdx.x;
    const float* p = kv + (size_t)b * L_ * DC_ + d;
    float s = 0.f;
    for (int l = 0; l < L_; ++l) s += p[(size_t)l * DC_];
    kvsum[b * DC_ + d] = s;
}

// GEMM 1: q_hk[64,16384] = hidden_q[64,2048] x q_proj_w[16384,2048]^T (TN)
__global__ void k_gemm_qhk(const float* __restrict__ hq, const float* __restrict__ qw,
                           float* __restrict__ qhk) {
    const int lane = threadIdx.x & 31;
    const int nt   = blockIdx.x * 8 + (threadIdx.x >> 5);   // 1024 tiles
    wave_gemm<4, true>(hq, DQ_, qw + (size_t)nt * 16 * DQ_, DQ_,
                       qhk + nt * 16, H_ * KH_, DQ_, lane);
}

// GEMM 2 (per head): q_big[:,h,:] = q_hk_h[64,128] x w_kc_q[h][128,1536] (NN)
__global__ void k_gemm_qbig(const float* __restrict__ qhk, const float* __restrict__ wq,
                            float* __restrict__ qbig) {
    const int lane = threadIdx.x & 31;
    const int h    = blockIdx.y;
    const int nt   = blockIdx.x * 8 + (threadIdx.x >> 5);   // 96 tiles
    wave_gemm<4, false>(qhk + h * KH_, H_ * KH_,
                        wq + (size_t)h * KH_ * DCQ_ + nt * 16, DCQ_,
                        qbig + (size_t)h * DCQ_ + nt * 16, H_ * DCQ_, KH_, lane);
}

// GEMM 3 (per head): q_r_h[64,64] = q_big_h[64,1536] x W_qr[h][1536,64] (NN)
__global__ void k_gemm_qr(const float* __restrict__ qbig, const float* __restrict__ wqr,
                          float* __restrict__ qr) {
    const int lane = threadIdx.x & 31;
    const int h    = blockIdx.y;
    const int nt   = threadIdx.x >> 5;                      // 4 tiles
    wave_gemm<4, false>(qbig + (size_t)h * DCQ_, H_ * DCQ_,
                        wqr + (size_t)h * DCQ_ * R_ + nt * 16, R_,
                        qr + h * R_ + nt * 16, H_ * R_, DCQ_, lane);
}

// GEMM 4 (per head): s1_h[64,512] = q_r_h[64,64] x W_kr[h][512,64]^T (TN)
__global__ void k_gemm_s1(const float* __restrict__ qr, const float* __restrict__ wkr,
                          float* __restrict__ s1) {
    const int lane = threadIdx.x & 31;
    const int h    = blockIdx.y;
    const int nt   = blockIdx.x * 8 + (threadIdx.x >> 5);   // 32 tiles
    wave_gemm<4, true>(qr + h * R_, H_ * R_,
                       wkr + (size_t)h * DC_ * R_ + (size_t)nt * 16 * R_, R_,
                       s1 + (size_t)h * DC_ + nt * 16, H_ * DC_, R_, lane);
}

// attn[b,h,:] = softmax(s1[b,h,:] * kv_sum[b,:])   (in-place over s1)
__global__ void k_softmax(float* __restrict__ s1, const float* __restrict__ kvsum) {
    __shared__ float red[256];
    const int bh = blockIdx.x;
    const int b  = bh >> 7;               // /H_
    const int t  = threadIdx.x;
    float* row = s1 + (size_t)bh * DC_;
    const float* ks = kvsum + b * DC_;
    float v0 = row[t]       * ks[t];
    float v1 = row[t + 256] * ks[t + 256];
    red[t] = fmaxf(v0, v1); __syncthreads();
    for (int s = 128; s > 0; s >>= 1) {
        if (t < s) red[t] = fmaxf(red[t], red[t + s]);
        __syncthreads();
    }
    const float m = red[0]; __syncthreads();
    float e0 = __expf(v0 - m), e1 = __expf(v1 - m);
    red[t] = e0 + e1; __syncthreads();
    for (int s = 128; s > 0; s >>= 1) {
        if (t < s) red[t] += red[t + s];
        __syncthreads();
    }
    const float rinv = 1.0f / red[0];
    row[t] = e0 * rinv;
    row[t + 256] = e1 * rinv;
}

// GEMM 5 (per batch): ctx_c[b] = attn[b][128,512] x kv[b][512,512] (NN, MT=8)
__global__ void k_gemm_ctx(const float* __restrict__ attn, const float* __restrict__ kv,
                           float* __restrict__ ctxc) {
    const int lane = threadIdx.x & 31;
    const int b    = blockIdx.y;
    const int nt   = blockIdx.x * 8 + (threadIdx.x >> 5);   // 32 tiles
    wave_gemm<8, false>(attn + (size_t)b * H_ * DC_, DC_,
                        kv + (size_t)b * L_ * DC_ + nt * 16, DC_,
                        ctxc + (size_t)b * H_ * DC_ + nt * 16, DC_, DC_, lane);
}

// GEMM 6 (per head): ctx_lat[:,h*128:] = ctx_c_h[64,512] x w_kc_kv[h][128,512]^T (TN)
__global__ void k_gemm_lat(const float* __restrict__ ctxc, const float* __restrict__ wkv,
                           float* __restrict__ lat) {
    const int lane = threadIdx.x & 31;
    const int h    = blockIdx.y;
    const int nt   = threadIdx.x >> 5;                      // 8 tiles
    wave_gemm<4, true>(ctxc + (size_t)h * DC_, H_ * DC_,
                       wkv + (size_t)h * KH_ * DC_ + (size_t)nt * 16 * DC_, DC_,
                       lat + h * KH_ + nt * 16, H_ * KH_, DC_, lane);
}

// GEMM 7: out[64,2048] = ctx_lat[64,16384] x out_proj_w[2048,16384]^T (TN)
__global__ void k_gemm_out(const float* __restrict__ lat, const float* __restrict__ ow,
                           float* __restrict__ out) {
    const int lane = threadIdx.x & 31;
    const int nt   = blockIdx.x * 8 + (threadIdx.x >> 5);   // 128 tiles
    wave_gemm<4, true>(lat, H_ * KH_, ow + (size_t)nt * 16 * (H_ * KH_), H_ * KH_,
                       out + nt * 16, DQ_, H_ * KH_, lane);
}

// ------------------------------- launcher -----------------------------------

extern "C" void kernel_launch(void* const* d_in, const int* in_sizes, int n_in,
                              void* d_out, int out_size, void* d_ws, size_t ws_size,
                              hipStream_t stream) {
    const float* hidden_q   = (const float*)d_in[0];
    const float* kv_c       = (const float*)d_in[1];
    const float* q_proj_w   = (const float*)d_in[2];
    const float* w_kc_q     = (const float*)d_in[3];
    const float* w_kc_kv    = (const float*)d_in[4];
    const float* W_qr       = (const float*)d_in[5];
    const float* W_kr       = (const float*)d_in[6];
    const float* out_proj_w = (const float*)d_in[7];
    float* out = (float*)d_out;

    float* ws = (float*)d_ws;
    float* kv     = ws;                                   // 16,777,216
    float* kvsum  = kv     + (size_t)B_ * L_ * DC_;       //     32,768
    float* qhk    = kvsum  + (size_t)B_ * DC_;            //  1,048,576
    float* qbig   = qhk    + (size_t)B_ * H_ * KH_;       // 12,582,912
    float* qr     = qbig   + (size_t)B_ * H_ * DCQ_;      //    524,288
    float* s1     = qr     + (size_t)B_ * H_ * R_;        //  4,194,304 (attn in-place)
    float* ctxc   = s1     + (size_t)B_ * H_ * DC_;       //  4,194,304
    float* lat    = ctxc   + (size_t)B_ * H_ * DC_;       //  1,048,576

    k_rope<<<dim3(B_ * L_), dim3(256), 0, stream>>>(kv_c, kv);
    k_kvsum<<<dim3(B_, 2), dim3(256), 0, stream>>>(kv, kvsum);
    k_gemm_qhk<<<dim3(128), dim3(256), 0, stream>>>(hidden_q, q_proj_w, qhk);
    k_gemm_qbig<<<dim3(12, H_), dim3(256), 0, stream>>>(qhk, w_kc_q, qbig);
    k_gemm_qr<<<dim3(1, H_), dim3(128), 0, stream>>>(qbig, W_qr, qr);
    k_gemm_s1<<<dim3(4, H_), dim3(256), 0, stream>>>(qr, W_kr, s1);
    k_softmax<<<dim3(B_ * H_), dim3(256), 0, stream>>>(s1, kvsum);
    k_gemm_ctx<<<dim3(4, B_), dim3(256), 0, stream>>>(s1, kv, ctxc);
    k_gemm_lat<<<dim3(1, H_), dim3(256), 0, stream>>>(ctxc, w_kc_kv, lat);
    k_gemm_out<<<dim3(16), dim3(256), 0, stream>>>(lat, out_proj_w, out);
}
